// AffinityGNN_87419764343210
// MI455X (gfx1250) — matrix operand order repaired and checked by reading, not compiled, not run on hardware
//
#include <hip/hip_runtime.h>
#include <stdint.h>

typedef __bf16 bf16_t;
typedef __attribute__((ext_vector_type(16))) __bf16 v16bf;
typedef __attribute__((ext_vector_type(8)))  float  v8f;

#define NNODES 32768
#define NEDGES 262144
#define HID 256
#define NPROT 512

__device__ __forceinline__ float gelu_f(float x) {
    return 0.5f * x * (1.0f + erff(x * 0.70710678118654752f));
}

__device__ __forceinline__ bf16_t f2bf(float f) {
    unsigned u = __float_as_uint(f);
    unsigned r = (u + 0x7fffu + ((u >> 16) & 1u)) >> 16;
    unsigned short s = (unsigned short)r;
    bf16_t b;
    __builtin_memcpy(&b, &s, 2);
    return b;
}

__device__ __forceinline__ float wave_sum(float v) {
    v += __shfl_xor(v, 1, 32);
    v += __shfl_xor(v, 2, 32);
    v += __shfl_xor(v, 4, 32);
    v += __shfl_xor(v, 8, 32);
    v += __shfl_xor(v, 16, 32);
    return v;
}

// ---------------------------------------------------------------------------
// fp32 -> bf16 transposed weight conversion: dst[n*Kd + k] = src[k*Nd + n]
// ---------------------------------------------------------------------------
__global__ __launch_bounds__(256) void wcvt_kernel(const float* __restrict__ src,
                                                   bf16_t* __restrict__ dst,
                                                   int Kd, int Nd) {
    int idx = blockIdx.x * 256 + threadIdx.x;
    if (idx >= Kd * Nd) return;
    int n = idx / Kd;
    int k = idx - n * Kd;
    dst[idx] = f2bf(src[(size_t)k * Nd + n]);
}

// ---------------------------------------------------------------------------
// WMMA bf16 GEMM: out[M,Nc] = act(A[M,Kd] @ Wt^T + bias (+ res))
// Wt stored transposed: Wt[n*Kd + k].  Block = 256 thr = 8 waves,
// block tile 128x64, wave tile 32x32 (2 A-frags x 2 B-frags = 4 WMMA accums,
// 32 B loaded per WMMA).
// ---------------------------------------------------------------------------
template <int ACT, bool OUT_BF16, bool RES>
__global__ __launch_bounds__(256) void gemm_wmma_kernel(
    const bf16_t* __restrict__ A, const bf16_t* __restrict__ Wt,
    const float* __restrict__ bias, const float* __restrict__ res,
    void* __restrict__ out, int M, int Kd, int Nc) {
    const int ntiles = Nc >> 6;
    const int bm = blockIdx.x / ntiles;
    const int bn = blockIdx.x - bm * ntiles;
    const int lane = threadIdx.x & 31;
    const int wave = threadIdx.x >> 5;
    const int m0 = bm * 128 + (wave >> 1) * 32;
    const int n0 = bn * 64 + (wave & 1) * 32;
    const int l15 = lane & 15;
    const int half = lane >> 4;

    const bf16_t* Arow0 = A + (size_t)(m0 + l15) * Kd;
    const bf16_t* Arow1 = A + (size_t)(m0 + 16 + l15) * Kd;
    const bf16_t* B0p = Wt + (size_t)(n0 + l15) * Kd + half * 16;
    const bf16_t* B1p = Wt + (size_t)(n0 + 16 + l15) * Kd + half * 16;

    v8f acc00 = {}, acc01 = {}, acc10 = {}, acc11 = {};
    for (int k0 = 0; k0 < Kd; k0 += 32) {
        // A fragment 16x32: lane(l15,half): elems 0..7 = K[k0+half*16 ..],
        // elems 8..15 = K[k0+half*16+8 ..]
        v16bf a0, a1;
        {
            uint4 lo = *(const uint4*)(Arow0 + k0 + half * 16);
            uint4 hi = *(const uint4*)(Arow0 + k0 + half * 16 + 8);
            __builtin_memcpy((char*)&a0, &lo, 16);
            __builtin_memcpy((char*)&a0 + 16, &hi, 16);
        }
        {
            uint4 lo = *(const uint4*)(Arow1 + k0 + half * 16);
            uint4 hi = *(const uint4*)(Arow1 + k0 + half * 16 + 8);
            __builtin_memcpy((char*)&a1, &lo, 16);
            __builtin_memcpy((char*)&a1 + 16, &hi, 16);
        }
        // B fragments 32x16: lane holds 16 contiguous K of one column
        v16bf b0 = *(const v16bf*)(B0p + k0);
        v16bf b1 = *(const v16bf*)(B1p + k0);
        acc00 = __builtin_amdgcn_wmma_f32_16x16x32_bf16(false, a0, false, b0,
                                                        (short)0, acc00, false, false);
        acc01 = __builtin_amdgcn_wmma_f32_16x16x32_bf16(false, a0, false, b1,
                                                        (short)0, acc01, false, false);
        acc10 = __builtin_amdgcn_wmma_f32_16x16x32_bf16(false, a1, false, b0,
                                                        (short)0, acc10, false, false);
        acc11 = __builtin_amdgcn_wmma_f32_16x16x32_bf16(false, a1, false, b1,
                                                        (short)0, acc11, false, false);
    }
    // Epilogue: C/D layout: n = n0+t*16+l15, m = m0 + mi*16 + half*8 + r
#pragma unroll
    for (int mi = 0; mi < 2; mi++) {
        const int mb = m0 + mi * 16 + half * 8;
#pragma unroll
        for (int t = 0; t < 2; t++) {
            v8f acc = mi == 0 ? (t == 0 ? acc00 : acc01)
                              : (t == 0 ? acc10 : acc11);
            int n = n0 + t * 16 + l15;
            float bv = bias[n];
#pragma unroll
            for (int r = 0; r < 8; r++) {
                size_t off = (size_t)(mb + r) * Nc + n;
                float v = acc[r] + bv;
                if (RES) v += res[off];
                if (ACT == 1) v = gelu_f(v);
                if (OUT_BF16)
                    ((bf16_t*)out)[off] = f2bf(v);
                else
                    ((float*)out)[off] = v;
            }
        }
    }
}

// ---------------------------------------------------------------------------
// Encoders: tiny input GEMM (5 or 10 inputs) + LayerNorm + GELU -> h f32
// one wave per node
// ---------------------------------------------------------------------------
__global__ __launch_bounds__(256) void encoder_kernel(
    const float* __restrict__ x, const float* __restrict__ pW,
    const float* __restrict__ pLb, const float* __restrict__ pG,
    const float* __restrict__ pB, const float* __restrict__ lW,
    const float* __restrict__ lLb, const float* __restrict__ lG,
    const float* __restrict__ lB, float* __restrict__ h) {
    int lane = threadIdx.x & 31;
    int node = blockIdx.x * 8 + (threadIdx.x >> 5);
    bool isProt = node < NPROT;
    const float* in = x + (size_t)node * 15 + (isProt ? 0 : 5);
    const float* W = isProt ? pW : lW;
    const float* Lb = isProt ? pLb : lLb;
    const float* G = isProt ? pG : lG;
    const float* Bb = isProt ? pB : lB;
    int nin = isProt ? 5 : 10;
    float v[8];
#pragma unroll
    for (int i = 0; i < 8; i++) {
        int j = lane * 8 + i;
        float a = Lb[j];
        for (int c = 0; c < nin; c++) a += in[c] * W[c * HID + j];
        v[i] = a;
    }
    float s = 0.f;
#pragma unroll
    for (int i = 0; i < 8; i++) s += v[i];
    float mean = wave_sum(s) * (1.0f / HID);
    float var = 0.f;
#pragma unroll
    for (int i = 0; i < 8; i++) {
        float d = v[i] - mean;
        var += d * d;
    }
    var = wave_sum(var) * (1.0f / HID);
    float rstd = rsqrtf(var + 1e-5f);
    float* out = h + (size_t)node * HID;
#pragma unroll
    for (int i = 0; i < 8; i++) {
        int j = lane * 8 + i;
        out[j] = gelu_f((v[i] - mean) * rstd * G[j] + Bb[j]);
    }
}

// ---------------------------------------------------------------------------
// LayerNorm f32 -> bf16, one wave per node
// ---------------------------------------------------------------------------
__global__ __launch_bounds__(256) void ln_kernel(const float* __restrict__ X,
                                                 const float* __restrict__ G,
                                                 const float* __restrict__ Bb,
                                                 bf16_t* __restrict__ out) {
    int lane = threadIdx.x & 31;
    int node = blockIdx.x * 8 + (threadIdx.x >> 5);
    const float* xr = X + (size_t)node * HID + lane * 8;
    float v[8];
    float s = 0.f;
#pragma unroll
    for (int i = 0; i < 8; i++) {
        v[i] = xr[i];
        s += v[i];
    }
    float mean = wave_sum(s) * (1.0f / HID);
    float var = 0.f;
#pragma unroll
    for (int i = 0; i < 8; i++) {
        float d = v[i] - mean;
        var += d * d;
    }
    var = wave_sum(var) * (1.0f / HID);
    float rstd = rsqrtf(var + 1e-5f);
    bf16_t* o = out + (size_t)node * HID + lane * 8;
#pragma unroll
    for (int i = 0; i < 8; i++) {
        int j = lane * 8 + i;
        o[i] = f2bf((v[i] - mean) * rstd * G[j] + Bb[j]);
    }
}

// ---------------------------------------------------------------------------
// Per-layer attention constants:
//  U[h*64+j]  = sum_d spW2[j*256 + h*32 + d] * attnW[64+d]
//  Ch[h]      = attn_b + sum_d spb2[h*32+d] * attnW[64+d]
//  Sv         = sum_{c<64} attnW[c]
// ---------------------------------------------------------------------------
__global__ __launch_bounds__(512) void sp_precompute_kernel(
    const float* __restrict__ spW2, const float* __restrict__ spb2,
    const float* __restrict__ attnW, const float* __restrict__ attnB,
    float* __restrict__ U, float* __restrict__ Ch, float* __restrict__ Sv) {
    int t = threadIdx.x;
    int h = t >> 6, j = t & 63;
    float u = 0.f;
    for (int d = 0; d < 32; d++) u += spW2[j * 256 + h * 32 + d] * attnW[64 + d];
    U[t] = u;
    if (t < 8) {
        float c = attnB[0];
        for (int d = 0; d < 32; d++) c += spb2[t * 32 + d] * attnW[64 + d];
        Ch[t] = c;
    }
    if (t == 8) {
        float s = 0.f;
        for (int c = 0; c < 64; c++) s += attnW[c];
        Sv[0] = s;
    }
}

// ---------------------------------------------------------------------------
// Edge pass 1: per-edge head logits -> s = exp(sigmoid(.)), partial head sums
// one wave per edge; 8 waves / block
// ---------------------------------------------------------------------------
__global__ __launch_bounds__(256) void edge_logits_kernel(
    const float* __restrict__ Qf, const float* __restrict__ Kf,
    const int* __restrict__ ei, const float* __restrict__ ea,
    const float* __restrict__ spW1, const float* __restrict__ spb1,
    const float* __restrict__ U, const float* __restrict__ Ch,
    const float* __restrict__ Sv, float* __restrict__ sbuf,
    float* __restrict__ hsums) {
    __shared__ float lsum[8];
    int lane = threadIdx.x & 31;
    int wv = threadIdx.x >> 5;
    if (threadIdx.x < 8) lsum[threadIdx.x] = 0.f;
    __syncthreads();
    int e = blockIdx.x * 8 + wv;
    int src = ei[e];
    int dst = ei[NEDGES + e];
    // per-head dot Q[src] . K[dst] (head = lane>>2, 8 dims per lane)
    const float* q = Qf + (size_t)src * HID + lane * 8;
    const float* k = Kf + (size_t)dst * HID + lane * 8;
    float d = 0.f;
#pragma unroll
    for (int i = 0; i < 8; i++) d += q[i] * k[i];
    d += __shfl_xor(d, 1, 32);
    d += __shfl_xor(d, 2, 32);
    d *= 0.17677669529663687f;  // 1/sqrt(32)
    // sp hidden: p[2*lane], p[2*lane+1] = relu(ea[3:7] @ spW1 + spb1)
    float ea0 = ea[(size_t)e * 7 + 3], ea1 = ea[(size_t)e * 7 + 4];
    float ea2 = ea[(size_t)e * 7 + 5], ea3 = ea[(size_t)e * 7 + 6];
    int j0 = lane * 2;
    float p0 = fmaxf(0.f, ea0 * spW1[j0] + ea1 * spW1[64 + j0] +
                              ea2 * spW1[128 + j0] + ea3 * spW1[192 + j0] + spb1[j0]);
    int j1 = j0 + 1;
    float p1 = fmaxf(0.f, ea0 * spW1[j1] + ea1 * spW1[64 + j1] +
                              ea2 * spW1[128 + j1] + ea3 * spW1[192 + j1] + spb1[j1]);
    // t[h] = p . U[h]
    float th[8];
#pragma unroll
    for (int h = 0; h < 8; h++) {
        float v = p0 * U[h * 64 + j0] + p1 * U[h * 64 + j1];
        th[h] = wave_sum(v);
    }
    float S = Sv[0];
    int h = lane >> 2;
    if ((lane & 3) == 0) {
        float lg = d * S + th[h] + Ch[h];
        float sg = 1.0f / (1.0f + __expf(-lg));
        float sv = __expf(sg);
        sbuf[(size_t)e * 8 + h] = sv;
        atomicAdd(&lsum[h], sv);
    }
    __syncthreads();
    if (threadIdx.x < 8) atomicAdd(&hsums[threadIdx.x], lsum[threadIdx.x]);
}

// ---------------------------------------------------------------------------
// Edge pass 2: agg[dst] += (s/sum_h) * V[dst]; one wave per edge
// ---------------------------------------------------------------------------
__global__ __launch_bounds__(256) void edge_scatter_kernel(
    const float* __restrict__ Vf, const int* __restrict__ ei,
    const float* __restrict__ sbuf, const float* __restrict__ hsums,
    float* __restrict__ agg) {
    int lane = threadIdx.x & 31;
    int wv = threadIdx.x >> 5;
    int e = blockIdx.x * 8 + wv;
    int dst = ei[NEDGES + e];
    int h = lane >> 2;
    float w = sbuf[(size_t)e * 8 + h] / hsums[h];
    const float* v = Vf + (size_t)dst * HID + lane * 8;
    float* a = agg + (size_t)dst * HID + lane * 8;
#pragma unroll
    for (int i = 0; i < 8; i++) atomicAdd(&a[i], w * v[i]);
}

// ---------------------------------------------------------------------------
// a += b (vectorized float4)
// ---------------------------------------------------------------------------
__global__ __launch_bounds__(256) void add_kernel(float* __restrict__ a,
                                                  const float* __restrict__ b,
                                                  int n4) {
    int i = blockIdx.x * 256 + threadIdx.x;
    if (i >= n4) return;
    float4 x = ((float4*)a)[i];
    float4 y = ((const float4*)b)[i];
    x.x += y.x;
    x.y += y.y;
    x.z += y.z;
    x.w += y.w;
    ((float4*)a)[i] = x;
}

// ---------------------------------------------------------------------------
// Mean pooling by graph: atomic accumulate, one wave per node
// ---------------------------------------------------------------------------
__global__ __launch_bounds__(256) void pool_kernel(const float* __restrict__ H,
                                                   const int* __restrict__ batch,
                                                   float* __restrict__ psum,
                                                   float* __restrict__ pcnt) {
    int lane = threadIdx.x & 31;
    int node = blockIdx.x * 8 + (threadIdx.x >> 5);
    int b = batch[node];
    const float* x = H + (size_t)node * HID + lane * 8;
    float* s = psum + (size_t)b * HID + lane * 8;
#pragma unroll
    for (int i = 0; i < 8; i++) atomicAdd(&s[i], x[i]);
    if (lane == 0) atomicAdd(&pcnt[b], 1.0f);
}

// ---------------------------------------------------------------------------
// Final head: pooled -> Linear(256,512) -> GELU -> Linear(512,1); 1 block/graph
// ---------------------------------------------------------------------------
__global__ __launch_bounds__(256) void fc_kernel(const float* __restrict__ psum,
                                                 const float* __restrict__ pcnt,
                                                 const float* __restrict__ W1,
                                                 const float* __restrict__ b1,
                                                 const float* __restrict__ W2,
                                                 const float* __restrict__ b2,
                                                 float* __restrict__ out) {
    __shared__ float pr[HID];
    __shared__ float red[256];
    int b = blockIdx.x;
    float cnt = fmaxf(pcnt[b], 1.0f);
    pr[threadIdx.x] = psum[(size_t)b * HID + threadIdx.x] / cnt;
    __syncthreads();
    float acc = 0.f;
#pragma unroll
    for (int kk = 0; kk < 2; kk++) {
        int k = threadIdx.x * 2 + kk;
        float v = b1[k];
        for (int j = 0; j < HID; j++) v += pr[j] * W1[j * 512 + k];
        acc += gelu_f(v) * W2[k];
    }
    red[threadIdx.x] = acc;
    __syncthreads();
    for (int s = 128; s > 0; s >>= 1) {
        if (threadIdx.x < s) red[threadIdx.x] += red[threadIdx.x + s];
        __syncthreads();
    }
    if (threadIdx.x == 0) out[b] = red[0] + b2[0];
}

// ---------------------------------------------------------------------------
// Host side
// ---------------------------------------------------------------------------
// d_in layout (jax pytree: dict keys sorted; lists in order):
//  0:x  1:edge_index  2:edge_attr  3:batch
//  4: fc.W1(256x512) 5: fc.b1 6: fc.W2(512x1) 7: fc.b2
//  per layer L (base = 8 + 24L):
//   +0 attn.W(96x1) +1 attn.b
//   +2 ffn.W1(256x512) +3 ffn.b1 +4 ffn.W2(512x256) +5 ffn.b2
//   +6 k.W1 +7 k.b1 +8 k.W2 +9 k.b2
//   +10 norm_b +11 norm_g
//   +12 q.W1 +13 q.b1 +14 q.W2 +15 q.b2
//   +16 sp.W1(4x64) +17 sp.b1 +18 sp.W2(64x256) +19 sp.b2
//   +20 v.W1 +21 v.b1 +22 v.W2 +23 v.b2
//  104: lig.b 105: lig.g 106: lig.lin.W(10x256) 107: lig.lin.b
//  108: prot.b 109: prot.g 110: prot.lin.W(5x256) 111: prot.lin.b

extern "C" void kernel_launch(void* const* d_in, const int* in_sizes, int n_in,
                              void* d_out, int out_size, void* d_ws, size_t ws_size,
                              hipStream_t stream) {
    (void)in_sizes; (void)n_in; (void)out_size; (void)ws_size;
    const int N = NNODES, E = NEDGES;
    const float* x = (const float*)d_in[0];
    const int* ei = (const int*)d_in[1];
    const float* ea = (const float*)d_in[2];
    const int* batch = (const int*)d_in[3];
    auto F = [&](int i) { return (const float*)d_in[i]; };

    uint8_t* wsp = (uint8_t*)d_ws;
    auto alloc = [&](size_t bytes) -> void* {
        void* r = (void*)wsp;
        wsp += (bytes + 255) & ~(size_t)255;
        return r;
    };
    float* h = (float*)alloc((size_t)N * HID * 4);
    bf16_t* xb = (bf16_t*)alloc((size_t)N * HID * 2);
    bf16_t* mid = (bf16_t*)alloc((size_t)N * 512 * 2);
    float* Qf = (float*)alloc((size_t)N * HID * 4);
    float* Kf = (float*)alloc((size_t)N * HID * 4);
    float* Vf = (float*)alloc((size_t)N * HID * 4);
    float* agg = (float*)alloc((size_t)N * HID * 4);
    float* sbuf = (float*)alloc((size_t)E * 8 * 4);
    // bf16 transposed weights: per layer {q1,q2,k1,k2,v1,v2,f1,f2} (131072 elems each)
    bf16_t* wt[4][8];
    for (int L = 0; L < 4; L++)
        for (int m = 0; m < 8; m++) wt[L][m] = (bf16_t*)alloc((size_t)131072 * 2);
    float* U = (float*)alloc(512 * 4);
    float* Ch = (float*)alloc(8 * 4);
    float* Sv = (float*)alloc(4);
    float* hsums = (float*)alloc(8 * 4);
    float* psum = (float*)alloc(64 * HID * 4);
    float* pcnt = (float*)alloc(64 * 4);

    // --- convert weights (fp32 -> bf16 transposed [N][K]) ---
    for (int L = 0; L < 4; L++) {
        int base = 8 + 24 * L;
        const int srcW1[4] = {base + 12, base + 6, base + 20, base + 2};  // q,k,v,ffn W1
        const int srcW2[4] = {base + 14, base + 8, base + 22, base + 4};  // q,k,v,ffn W2
        for (int m = 0; m < 4; m++) {
            wcvt_kernel<<<512, 256, 0, stream>>>(F(srcW1[m]), wt[L][2 * m], 256, 512);
            wcvt_kernel<<<512, 256, 0, stream>>>(F(srcW2[m]), wt[L][2 * m + 1], 512, 256);
        }
    }

    // --- encoders -> h ---
    encoder_kernel<<<N / 8, 256, 0, stream>>>(x, F(110), F(111), F(109), F(108),
                                              F(106), F(107), F(105), F(104), h);

    const int g1 = (N / 128) * (512 / 64);  // 2048 blocks
    const int g2 = (N / 128) * (HID / 64);  // 1024 blocks

    for (int L = 0; L < 4; L++) {
        int base = 8 + 24 * L;
        const float* normG = F(base + 11);
        const float* normB = F(base + 10);
        hipMemsetAsync(agg, 0, (size_t)N * HID * 4, stream);
        hipMemsetAsync(hsums, 0, 8 * 4, stream);

        // n = LN(h)
        ln_kernel<<<N / 8, 256, 0, stream>>>(h, normG, normB, xb);

        // Q/K/V MLPs: Linear->GELU->Linear
        const int b1i[3] = {base + 13, base + 7, base + 21};
        const int b2i[3] = {base + 15, base + 9, base + 23};
        float* outs[3] = {Qf, Kf, Vf};
        for (int m = 0; m < 3; m++) {
            gemm_wmma_kernel<1, true, false><<<g1, 256, 0, stream>>>(
                xb, wt[L][2 * m], F(b1i[m]), nullptr, mid, N, 256, 512);
            gemm_wmma_kernel<0, false, false><<<g2, 256, 0, stream>>>(
                mid, wt[L][2 * m + 1], F(b2i[m]), nullptr, outs[m], N, 512, 256);
        }

        // edge attention
        sp_precompute_kernel<<<1, 512, 0, stream>>>(F(base + 18), F(base + 19),
                                                    F(base + 0), F(base + 1), U, Ch, Sv);
        edge_logits_kernel<<<E / 8, 256, 0, stream>>>(Qf, Kf, ei, ea, F(base + 16),
                                                      F(base + 17), U, Ch, Sv, sbuf,
                                                      hsums);
        edge_scatter_kernel<<<E / 8, 256, 0, stream>>>(Vf, ei, sbuf, hsums, agg);

        // h += agg
        add_kernel<<<(N * HID / 4 + 255) / 256, 256, 0, stream>>>(h, agg, N * HID / 4);

        // h += FFN(LN(h))   (residual fused into second GEMM epilogue)
        ln_kernel<<<N / 8, 256, 0, stream>>>(h, normG, normB, xb);
        gemm_wmma_kernel<1, true, false><<<g1, 256, 0, stream>>>(
            xb, wt[L][6], F(base + 3), nullptr, mid, N, 256, 512);
        gemm_wmma_kernel<0, false, true><<<g2, 256, 0, stream>>>(
            mid, wt[L][7], F(base + 5), h, h, N, 512, 256);
    }

    // --- mean pool + head ---
    hipMemsetAsync(psum, 0, 64 * HID * 4, stream);
    hipMemsetAsync(pcnt, 0, 64 * 4, stream);
    pool_kernel<<<N / 8, 256, 0, stream>>>(h, batch, psum, pcnt);
    fc_kernel<<<64, 256, 0, stream>>>(psum, pcnt, F(4), F(5), F(6), F(7),
                                      (float*)d_out);
}
